// PCT_Cap_20529943675335
// MI455X (gfx1250) — matrix operand order repaired and checked by reading, not compiled, not run on hardware
//
#include <hip/hip_runtime.h>
#include <hip/hip_bf16.h>

// ---------------------------------------------------------------------------
// PCT-style network on gfx1250: all GEMMs via v_wmma_f32_16x16x32_f16.
// Shapes: B=8, N=2048, C=3, DE=256, NSA=4.  M = B*N = 16384.
// Every GEMM consumes B in [N,K] (transposed) storage so all fragment loads
// are contiguous 32-byte runs per lane -> paired global_load_b128.
// ---------------------------------------------------------------------------

typedef __attribute__((ext_vector_type(16))) _Float16 v16h;
typedef __attribute__((ext_vector_type(8)))  float    v8f;
typedef _Float16 half_t;

#define BB 8
#define NN 2048
#define MM (BB * NN)   // 16384
#define DE 256
#define NSA 4

// ---------------------------------------------------------------------------
// WMMA fragment loaders (wave32; A/B in contiguous k order, k = 16*(lane/16) + s)
// ---------------------------------------------------------------------------

// A 16x32 f16 fragment from per-lane base pointer p = &A[row, kb]:
// per lane, 16 contiguous halves: p[0..15]  (one 32B run).
__device__ inline v16h load_a16(const half_t* __restrict__ p) {
  v16h a;
#pragma unroll
  for (int j = 0; j < 8; ++j) {
    a[j]     = p[j];
    a[8 + j] = p[8 + j];
  }
  return a;
}

// B 32x16 f16 fragment from per-lane base pointer p = &Bt[col, kb]:
// per lane, 16 contiguous halves (one 32B run).
__device__ inline v16h load_b16(const half_t* __restrict__ p) {
  v16h b;
#pragma unroll
  for (int j = 0; j < 16; ++j) b[j] = p[j];
  return b;
}

// C/D 16x16 f32 accumulator: VGPR r -> row = row0 + r (lanes 0-15) or
// row0 + 8 + r (lanes 16-31); col = col0 + (lane&15).
__device__ inline void store_c_f32(float* __restrict__ C, int ldc, int row0,
                                   int col0, const v8f& c, const float* bias) {
  int lane = threadIdx.x & 31;
  int col  = col0 + (lane & 15);
  int rb   = row0 + ((lane & 16) ? 8 : 0);
  float bv = bias ? bias[col] : 0.0f;
#pragma unroll
  for (int r = 0; r < 8; ++r) C[(size_t)(rb + r) * ldc + col] = c[r] + bv;
}

__device__ inline void store_c_f16(half_t* __restrict__ C, int ldc, int row0,
                                   int col0, const v8f& c) {
  int lane = threadIdx.x & 31;
  int col  = col0 + (lane & 15);
  int rb   = row0 + ((lane & 16) ? 8 : 0);
#pragma unroll
  for (int r = 0; r < 8; ++r) C[(size_t)(rb + r) * ldc + col] = (half_t)c[r];
}

__device__ inline v8f load_c_f16(const half_t* __restrict__ C, int ldc,
                                 int row0, int col0) {
  int lane = threadIdx.x & 31;
  int col  = col0 + (lane & 15);
  int rb   = row0 + ((lane & 16) ? 8 : 0);
  v8f c;
#pragma unroll
  for (int r = 0; r < 8; ++r) c[r] = (float)C[(size_t)(rb + r) * ldc + col];
  return c;
}

#define WMMA_F16(a, b, c) \
  __builtin_amdgcn_wmma_f32_16x16x32_f16(false, (a), false, (b), (short)0, \
                                         (c), false, false)

// ---------------------------------------------------------------------------
// WMMA GEMM kernels. Block = 256 threads = 8 waves arranged 2x4, each wave
// owns a 32x32 register tile (2 A-frags x 2 B-frags, 4 WMMA per 32-K step).
// Block tile: 64 (M) x 128 (N). M % 64 == 0, N % 128 == 0, K % 32 == 0.
// A is [M,K] row-major (lda), Bt is [N,K] row-major (ldb).
// ---------------------------------------------------------------------------

// C(f32)[M,N] = A x Bt^T (+ optional per-column bias).
__global__ void wmma_gemm_f32(const half_t* __restrict__ A, int lda,
                              const half_t* __restrict__ Bt, int ldb,
                              const float* __restrict__ bias,
                              float* __restrict__ C, int ldc, int K) {
  int wid  = threadIdx.x >> 5;
  int lane = threadIdx.x & 31;
  int row0 = blockIdx.y * 64 + (wid >> 2) * 32;
  int col0 = blockIdx.x * 128 + (wid & 3) * 32;
  const half_t* pa0 = A + (size_t)(row0 + (lane & 15)) * lda
                        + ((lane & 16) ? 16 : 0);
  const half_t* pa1 = pa0 + (size_t)16 * lda;
  const half_t* pb0 = Bt + (size_t)(col0 + (lane & 15)) * ldb
                         + ((lane & 16) ? 16 : 0);
  const half_t* pb1 = pb0 + (size_t)16 * ldb;
  v8f c00 = {}, c01 = {}, c10 = {}, c11 = {};
  for (int k = 0; k < K; k += 32) {
    v16h a0 = load_a16(pa0), a1 = load_a16(pa1);
    v16h b0 = load_b16(pb0), b1 = load_b16(pb1);
    c00 = WMMA_F16(a0, b0, c00);
    c01 = WMMA_F16(a0, b1, c01);
    c10 = WMMA_F16(a1, b0, c10);
    c11 = WMMA_F16(a1, b1, c11);
    pa0 += 32; pa1 += 32; pb0 += 32; pb1 += 32;
  }
  store_c_f32(C, ldc, row0,      col0,      c00, bias);
  store_c_f32(C, ldc, row0,      col0 + 16, c01, bias);
  store_c_f32(C, ldc, row0 + 16, col0,      c10, bias);
  store_c_f32(C, ldc, row0 + 16, col0 + 16, c11, bias);
}

// C(f16)[M,N] = A x Bt^T.
__global__ void wmma_gemm_f16(const half_t* __restrict__ A, int lda,
                              const half_t* __restrict__ Bt, int ldb,
                              half_t* __restrict__ C, int ldc, int K) {
  int wid  = threadIdx.x >> 5;
  int lane = threadIdx.x & 31;
  int row0 = blockIdx.y * 64 + (wid >> 2) * 32;
  int col0 = blockIdx.x * 128 + (wid & 3) * 32;
  const half_t* pa0 = A + (size_t)(row0 + (lane & 15)) * lda
                        + ((lane & 16) ? 16 : 0);
  const half_t* pa1 = pa0 + (size_t)16 * lda;
  const half_t* pb0 = Bt + (size_t)(col0 + (lane & 15)) * ldb
                         + ((lane & 16) ? 16 : 0);
  const half_t* pb1 = pb0 + (size_t)16 * ldb;
  v8f c00 = {}, c01 = {}, c10 = {}, c11 = {};
  for (int k = 0; k < K; k += 32) {
    v16h a0 = load_a16(pa0), a1 = load_a16(pa1);
    v16h b0 = load_b16(pb0), b1 = load_b16(pb1);
    c00 = WMMA_F16(a0, b0, c00);
    c01 = WMMA_F16(a0, b1, c01);
    c10 = WMMA_F16(a1, b0, c10);
    c11 = WMMA_F16(a1, b1, c11);
    pa0 += 32; pa1 += 32; pb0 += 32; pb1 += 32;
  }
  store_c_f16(C, ldc, row0,      col0,      c00);
  store_c_f16(C, ldc, row0,      col0 + 16, c01);
  store_c_f16(C, ldc, row0 + 16, col0,      c10);
  store_c_f16(C, ldc, row0 + 16, col0 + 16, c11);
}

// Residual: C(f16)[M,N] += A x Bt^T.  C preloaded into the accumulators.
__global__ void wmma_gemm_res_f16(const half_t* __restrict__ A, int lda,
                                  const half_t* __restrict__ Bt, int ldb,
                                  half_t* __restrict__ C, int ldc, int K) {
  int wid  = threadIdx.x >> 5;
  int lane = threadIdx.x & 31;
  int row0 = blockIdx.y * 64 + (wid >> 2) * 32;
  int col0 = blockIdx.x * 128 + (wid & 3) * 32;
  const half_t* pa0 = A + (size_t)(row0 + (lane & 15)) * lda
                        + ((lane & 16) ? 16 : 0);
  const half_t* pa1 = pa0 + (size_t)16 * lda;
  const half_t* pb0 = Bt + (size_t)(col0 + (lane & 15)) * ldb
                         + ((lane & 16) ? 16 : 0);
  const half_t* pb1 = pb0 + (size_t)16 * ldb;
  v8f c00 = load_c_f16(C, ldc, row0,      col0);
  v8f c01 = load_c_f16(C, ldc, row0,      col0 + 16);
  v8f c10 = load_c_f16(C, ldc, row0 + 16, col0);
  v8f c11 = load_c_f16(C, ldc, row0 + 16, col0 + 16);
  for (int k = 0; k < K; k += 32) {
    v16h a0 = load_a16(pa0), a1 = load_a16(pa1);
    v16h b0 = load_b16(pb0), b1 = load_b16(pb1);
    c00 = WMMA_F16(a0, b0, c00);
    c01 = WMMA_F16(a0, b1, c01);
    c10 = WMMA_F16(a1, b0, c10);
    c11 = WMMA_F16(a1, b1, c11);
    pa0 += 32; pa1 += 32; pb0 += 32; pb1 += 32;
  }
  store_c_f16(C, ldc, row0,      col0,      c00);
  store_c_f16(C, ldc, row0,      col0 + 16, c01);
  store_c_f16(C, ldc, row0 + 16, col0,      c10);
  store_c_f16(C, ldc, row0 + 16, col0 + 16, c11);
}

// ---------------------------------------------------------------------------
// Pointwise / reduction kernels
// ---------------------------------------------------------------------------

__global__ void cast_f32_to_f16(const float* __restrict__ src,
                                half_t* __restrict__ dst, int n) {
  int i = blockIdx.x * blockDim.x + threadIdx.x;
  if (i < n) dst[i] = (half_t)src[i];
}

// conv1: h1[o, m] = sum_c x[m,c] * w1[o,c] + b1[o]   (channel-major out)
__global__ void conv1_kernel(const float* __restrict__ x,
                             const float* __restrict__ w1,
                             const float* __restrict__ b1,
                             float* __restrict__ h1) {
  int idx = blockIdx.x * blockDim.x + threadIdx.x;  // 64*MM threads
  if (idx >= 64 * MM) return;
  int o = idx >> 14;          // / MM
  int m = idx & (MM - 1);
  const float* xp = x + (size_t)m * 3;
  const float* wp = w1 + o * 3;
  h1[idx] = fmaf(xp[2], wp[2], fmaf(xp[1], wp[1], fmaf(xp[0], wp[0], b1[o])));
}

// BN stats over rows: X channel-major [Cch, M]; one block per channel.
__global__ void bn_stats_row(const float* __restrict__ X, int M,
                             float* __restrict__ mean,
                             float* __restrict__ rstd) {
  int c = blockIdx.x;
  const float* row = X + (size_t)c * M;
  float s = 0.f, s2 = 0.f;
  for (int i = threadIdx.x; i < M; i += 256) {
    float v = row[i];
    s += v; s2 += v * v;
  }
  __shared__ float ss[256], sq[256];
  ss[threadIdx.x] = s; sq[threadIdx.x] = s2;
  __syncthreads();
  for (int off = 128; off > 0; off >>= 1) {
    if (threadIdx.x < off) {
      ss[threadIdx.x] += ss[threadIdx.x + off];
      sq[threadIdx.x] += sq[threadIdx.x + off];
    }
    __syncthreads();
  }
  if (threadIdx.x == 0) {
    float mu  = ss[0] / (float)M;
    float var = sq[0] / (float)M - mu * mu;
    mean[c] = mu;
    rstd[c] = rsqrtf(var + 1e-5f);
  }
}

// BN stats over columns: X row-major [M, Cch]; one block per channel (col).
__global__ void bn_stats_col(const float* __restrict__ X, int M, int ld,
                             float* __restrict__ mean,
                             float* __restrict__ rstd) {
  int c = blockIdx.x;
  float s = 0.f, s2 = 0.f;
  for (int i = threadIdx.x; i < M; i += 256) {
    float v = X[(size_t)i * ld + c];
    s += v; s2 += v * v;
  }
  __shared__ float ss[256], sq[256];
  ss[threadIdx.x] = s; sq[threadIdx.x] = s2;
  __syncthreads();
  for (int off = 128; off > 0; off >>= 1) {
    if (threadIdx.x < off) {
      ss[threadIdx.x] += ss[threadIdx.x + off];
      sq[threadIdx.x] += sq[threadIdx.x + off];
    }
    __syncthreads();
  }
  if (threadIdx.x == 0) {
    float mu  = ss[0] / (float)M;
    float var = sq[0] / (float)M - mu * mu;
    mean[c] = mu;
    rstd[c] = rsqrtf(var + 1e-5f);
  }
}

// BN+ReLU on channel-major h1[64, M], transpose-store f16 A[m, 64].
__global__ void bn1_relu_cast(const float* __restrict__ h1,
                              const float* __restrict__ mean,
                              const float* __restrict__ rstd,
                              const float* __restrict__ g,
                              const float* __restrict__ be,
                              half_t* __restrict__ A) {
  int idx = blockIdx.x * blockDim.x + threadIdx.x;  // 64*MM
  if (idx >= 64 * MM) return;
  int m = idx >> 6;
  int o = idx & 63;
  float v = (h1[(size_t)o * MM + m] - mean[o]) * rstd[o] * g[o] + be[o];
  A[idx] = (half_t)fmaxf(v, 0.0f);
}

// BN+ReLU on row-major out[M, 256] -> f16 H[M, 256].
__global__ void bn2_relu_cast(const float* __restrict__ X,
                              const float* __restrict__ mean,
                              const float* __restrict__ rstd,
                              const float* __restrict__ g,
                              const float* __restrict__ be,
                              half_t* __restrict__ H) {
  int idx = blockIdx.x * blockDim.x + threadIdx.x;  // MM*256
  if (idx >= MM * DE) return;
  int e = idx & (DE - 1);
  float v = (X[idx] - mean[e]) * rstd[e] * g[e] + be[e];
  H[idx] = (half_t)fmaxf(v, 0.0f);
}

// Transpose V [BB*NN, DE] -> VT [BB, DE, NN]  (per batch: VT[e,n] = V[n,e])
__global__ void v_transpose(const half_t* __restrict__ V,
                            half_t* __restrict__ VT) {
  int idx = blockIdx.x * blockDim.x + threadIdx.x;  // MM*DE
  if (idx >= MM * DE) return;
  int m = idx >> 8;           // / DE
  int e = idx & (DE - 1);
  int b = m >> 11;            // / NN
  int n = m & (NN - 1);
  VT[((size_t)b * DE + e) * NN + n] = V[idx];
}

// Row softmax in place on S [rows x Nc] fp32. One block per row.
__global__ void softmax_row(float* __restrict__ S, int Nc) {
  float* row = S + (size_t)blockIdx.x * Nc;
  __shared__ float red[256];
  float mx = -3.0e38f;
  for (int i = threadIdx.x; i < Nc; i += 256) mx = fmaxf(mx, row[i]);
  red[threadIdx.x] = mx;
  __syncthreads();
  for (int off = 128; off > 0; off >>= 1) {
    if (threadIdx.x < off)
      red[threadIdx.x] = fmaxf(red[threadIdx.x], red[threadIdx.x + off]);
    __syncthreads();
  }
  mx = red[0];
  __syncthreads();
  float s = 0.f;
  for (int i = threadIdx.x; i < Nc; i += 256) {
    float e = __expf(row[i] - mx);
    row[i] = e;
    s += e;
  }
  red[threadIdx.x] = s;
  __syncthreads();
  for (int off = 128; off > 0; off >>= 1) {
    if (threadIdx.x < off) red[threadIdx.x] += red[threadIdx.x + off];
    __syncthreads();
  }
  float inv = 1.0f / red[0];
  for (int i = threadIdx.x; i < Nc; i += 256) row[i] *= inv;
}

// Column sums of S [Nc x Nc]: colsum[m] = sum_n S[n, m].
__global__ void colsum_kernel(const float* __restrict__ S,
                              float* __restrict__ colsum, int Nc) {
  int m = blockIdx.x * blockDim.x + threadIdx.x;
  if (m >= Nc) return;
  float s = 0.f;
  for (int n = 0; n < Nc; ++n) s += S[(size_t)n * Nc + m];
  colsum[m] = s;
}

// L1 column renorm + transpose to f16: attnT[m, n] = S[n, m] / (1e-9 + colsum[m])
__global__ void renorm_transpose(const float* __restrict__ S,
                                 const float* __restrict__ colsum,
                                 half_t* __restrict__ attnT, int Nc) {
  int idx = blockIdx.x * blockDim.x + threadIdx.x;  // Nc*Nc
  if (idx >= Nc * Nc) return;
  int n = idx / Nc;
  int m = idx - n * Nc;
  float v = S[idx] / (1e-9f + colsum[m]);
  attnT[(size_t)m * Nc + n] = (half_t)v;
}

// pooled[m] = mean_e H[m, e]
__global__ void pool_kernel(const half_t* __restrict__ H,
                            float* __restrict__ pooled) {
  int m = blockIdx.x * blockDim.x + threadIdx.x;
  if (m >= MM) return;
  const half_t* row = H + (size_t)m * DE;
  float s = 0.f;
  for (int e = 0; e < DE; ++e) s += (float)row[e];
  pooled[m] = s * (1.0f / DE);
}

// f1[b, j] = relu(sum_n pooled[b, n] * fc1_w[j, n] + fc1_b[j])
__global__ void fc1_kernel(const float* __restrict__ pooled,
                           const float* __restrict__ w,
                           const float* __restrict__ bias,
                           float* __restrict__ f1) {
  int idx = blockIdx.x * blockDim.x + threadIdx.x;  // BB*NN
  if (idx >= BB * NN) return;
  int b = idx / NN;
  int j = idx - b * NN;
  const float* pr = pooled + (size_t)b * NN;
  const float* wr = w + (size_t)j * NN;
  float acc = bias[j];
  for (int n = 0; n < NN; ++n) acc = fmaf(pr[n], wr[n], acc);
  f1[idx] = fmaxf(acc, 0.0f);
}

// out[b] = sum_j f1[b, j] * fc2_w[j] + fc2_b[0]
__global__ void fc2_kernel(const float* __restrict__ f1,
                           const float* __restrict__ w,
                           const float* __restrict__ bias,
                           float* __restrict__ out) {
  int b = blockIdx.x;
  const float* fr = f1 + (size_t)b * NN;
  float s = 0.f;
  for (int j = threadIdx.x; j < NN; j += 256) s = fmaf(fr[j], w[j], s);
  __shared__ float red[256];
  red[threadIdx.x] = s;
  __syncthreads();
  for (int off = 128; off > 0; off >>= 1) {
    if (threadIdx.x < off) red[threadIdx.x] += red[threadIdx.x + off];
    __syncthreads();
  }
  if (threadIdx.x == 0) out[b] = red[0] + bias[0];
}

// ---------------------------------------------------------------------------
// Host launcher
// ---------------------------------------------------------------------------

extern "C" void kernel_launch(void* const* d_in, const int* in_sizes, int n_in,
                              void* d_out, int out_size, void* d_ws,
                              size_t ws_size, hipStream_t stream) {
  (void)in_sizes; (void)n_in; (void)out_size; (void)ws_size;

  const float* x     = (const float*)d_in[0];
  const float* w1    = (const float*)d_in[1];
  const float* b1    = (const float*)d_in[2];
  const float* g1    = (const float*)d_in[3];
  const float* be1   = (const float*)d_in[4];
  const float* w2    = (const float*)d_in[5];
  const float* b2    = (const float*)d_in[6];
  const float* g2    = (const float*)d_in[7];
  const float* be2   = (const float*)d_in[8];
  const float* wq    = (const float*)d_in[9];
  const float* wk    = (const float*)d_in[10];
  const float* wv    = (const float*)d_in[11];
  const float* fc1_w = (const float*)d_in[12];
  const float* fc1_b = (const float*)d_in[13];
  const float* fc2_w = (const float*)d_in[14];
  const float* fc2_b = (const float*)d_in[15];
  float* out = (float*)d_out;

  // Bump-allocate workspace (all chunks 256B-aligned).
  char* p = (char*)d_ws;
  auto alloc = [&](size_t bytes) {
    void* r = (void*)p;
    p += (bytes + 255) & ~(size_t)255;
    return r;
  };
  float*  h1     = (float*)alloc((size_t)64 * MM * 4);          // [64, M]
  float*  mean1  = (float*)alloc(64 * 4);
  float*  rstd1  = (float*)alloc(64 * 4);
  half_t* A64    = (half_t*)alloc((size_t)MM * 64 * 2);         // [M, 64]
  float*  o256   = (float*)alloc((size_t)MM * DE * 4);          // [M, 256]
  float*  mean2  = (float*)alloc(DE * 4);
  float*  rstd2  = (float*)alloc(DE * 4);
  half_t* H      = (half_t*)alloc((size_t)MM * DE * 2);         // [M, 256]
  half_t* w2h    = (half_t*)alloc((size_t)DE * 64 * 2);
  half_t* wqh    = (half_t*)alloc((size_t)NSA * DE * DE * 2);
  half_t* wkh    = (half_t*)alloc((size_t)NSA * DE * DE * 2);
  half_t* wvh    = (half_t*)alloc((size_t)NSA * DE * DE * 2);
  half_t* Q      = (half_t*)alloc((size_t)MM * DE * 2);         // [M, 256]
  half_t* Kb     = (half_t*)alloc((size_t)MM * DE * 2);         // [M, 256]
  half_t* V      = (half_t*)alloc((size_t)MM * DE * 2);         // [M, 256]
  half_t* VT     = (half_t*)alloc((size_t)BB * DE * NN * 2);    // [B, 256, N]
  float*  S      = (float*)alloc((size_t)NN * NN * 4);          // per-batch
  float*  csum   = (float*)alloc(NN * 4);
  half_t* attnT  = (half_t*)alloc((size_t)NN * NN * 2);         // per-batch
  float*  pooled = (float*)alloc((size_t)MM * 4);
  float*  f1     = (float*)alloc((size_t)BB * NN * 4);

  // ---- weight casts to f16 ----
  cast_f32_to_f16<<<(DE * 64 + 255) / 256, 256, 0, stream>>>(w2, w2h, DE * 64);
  int wsz = NSA * DE * DE;
  cast_f32_to_f16<<<(wsz + 255) / 256, 256, 0, stream>>>(wq, wqh, wsz);
  cast_f32_to_f16<<<(wsz + 255) / 256, 256, 0, stream>>>(wk, wkh, wsz);
  cast_f32_to_f16<<<(wsz + 255) / 256, 256, 0, stream>>>(wv, wvh, wsz);

  // ---- conv1 + BN1 + ReLU ----
  conv1_kernel<<<(64 * MM + 255) / 256, 256, 0, stream>>>(x, w1, b1, h1);
  bn_stats_row<<<64, 256, 0, stream>>>(h1, MM, mean1, rstd1);
  bn1_relu_cast<<<(64 * MM + 255) / 256, 256, 0, stream>>>(h1, mean1, rstd1,
                                                           g1, be1, A64);

  // ---- conv2 (WMMA GEMM, M=16384, N=256, K=64) + BN2 + ReLU ----
  {
    dim3 grid(DE / 128, MM / 64);
    wmma_gemm_f32<<<grid, 256, 0, stream>>>(A64, 64, w2h, 64, b2, o256, DE,
                                            64);
  }
  bn_stats_col<<<DE, 256, 0, stream>>>(o256, MM, DE, mean2, rstd2);
  bn2_relu_cast<<<(MM * DE + 255) / 256, 256, 0, stream>>>(o256, mean2, rstd2,
                                                           g2, be2, H);

  // ---- NSA offset-attention layers ----
  for (int i = 0; i < NSA; ++i) {
    const half_t* wqi = wqh + (size_t)i * DE * DE;
    const half_t* wki = wkh + (size_t)i * DE * DE;
    const half_t* wvi = wvh + (size_t)i * DE * DE;

    // Q/K/V projections: [16384,256] = H[16384,256] x W^T (W stored [out,in]).
    dim3 gproj(DE / 128, MM / 64);
    wmma_gemm_f16<<<gproj, 256, 0, stream>>>(H, DE, wqi, DE, Q, DE, DE);
    wmma_gemm_f16<<<gproj, 256, 0, stream>>>(H, DE, wki, DE, Kb, DE, DE);
    wmma_gemm_f16<<<gproj, 256, 0, stream>>>(H, DE, wvi, DE, V, DE, DE);
    v_transpose<<<(MM * DE + 255) / 256, 256, 0, stream>>>(V, VT);

    for (int b = 0; b < BB; ++b) {
      const half_t* Qb  = Q  + (size_t)b * NN * DE;
      const half_t* Kbb = Kb + (size_t)b * NN * DE;
      const half_t* VTb = VT + (size_t)b * DE * NN;
      half_t* Hb = H + (size_t)b * NN * DE;

      // S[n, m] = sum_a Q[n, a] * K[m, a]   (K-matrix already [N,K] layout)
      dim3 glog(NN / 128, NN / 64);
      wmma_gemm_f32<<<glog, 256, 0, stream>>>(Qb, DE, Kbb, DE, nullptr, S, NN,
                                              DE);

      softmax_row<<<NN, 256, 0, stream>>>(S, NN);
      colsum_kernel<<<NN / 256, 256, 0, stream>>>(S, csum, NN);
      renorm_transpose<<<(NN * NN + 255) / 256, 256, 0, stream>>>(S, csum,
                                                                  attnT, NN);

      // H[m, e] += sum_n attnT[m, n] * VT[e, n]   (WMMA, K=2048, residual)
      dim3 gres(DE / 128, NN / 64);
      wmma_gemm_res_f16<<<gres, 256, 0, stream>>>(attnT, NN, VTb, NN, Hb, DE,
                                                  NN);
    }
  }

  // ---- head: channel mean pool, fc1 + ReLU, fc2 ----
  pool_kernel<<<(MM + 255) / 256, 256, 0, stream>>>(H, pooled);
  fc1_kernel<<<(BB * NN + 255) / 256, 256, 0, stream>>>(pooled, fc1_w, fc1_b,
                                                        f1);
  fc2_kernel<<<BB, 256, 0, stream>>>(f1, fc2_w, fc2_b, out);
}
